// NonlearnableCell_35751307772219
// MI455X (gfx1250) — matrix operand order repaired and checked by reading, not compiled, not run on hardware
//
#include <hip/hip_runtime.h>
#include <hip/hip_bf16.h>

#define B_DIM 64
#define T_DIM 1024
#define IN_DIM 512
#define H_DIM 512
#define G_DIM 1536

typedef __attribute__((ext_vector_type(16))) __bf16 v16bf;
typedef __attribute__((ext_vector_type(8)))  float  v8f;

union Frag {
    v16bf v;
    uint4 q[2];
};

__device__ __forceinline__ unsigned short f2bf(float f) {
    unsigned u = __float_as_uint(f);
    unsigned r = 0x7FFFu + ((u >> 16) & 1u);   // round-to-nearest-even
    return (unsigned short)((u + r) >> 16);
}

// ---------------------------------------------------------------------------
// Kernel 1: ih_all[b,t,g] = xs[b,t,:] @ w_ih[b,:,g] + b_ih[b,g]
// 64x64 tile per 256-thread block; 8 waves (4 M-tiles x 2 N-halves);
// bf16 WMMA 16x16x32 with f32 accumulate; K staged through LDS.
// LDS row pitch = 40 bf16 (80 B) -> 16B aligned, conflict-free frag loads.
// ---------------------------------------------------------------------------
__global__ __launch_bounds__(256) void gru_ih_gemm(
    const float* __restrict__ xs, const float* __restrict__ w_ih,
    const float* __restrict__ b_ih, float* __restrict__ ih_all)
{
    __shared__ unsigned short As[64][40];  // [t-row][k]
    __shared__ unsigned short Bs[64][40];  // [g-col][k]  (transposed: K contiguous)

    const int b    = blockIdx.z;
    const int t0   = blockIdx.y * 64;
    const int g0   = blockIdx.x * 64;
    const int tid  = threadIdx.x;
    const int wid  = tid >> 5;
    const int lane = tid & 31;
    const int wm   = wid >> 1;     // 0..3 : 16-row M tile
    const int wn   = wid & 1;      // 0..1 : 32-col N half
    const int kh   = lane >> 4;    // lane half (K split per WMMA layout)
    const int ln   = lane & 15;

    const float* xsB = xs   + ((size_t)b * T_DIM + t0) * IN_DIM;
    const float* wB  = w_ih + (size_t)b * IN_DIM * G_DIM + g0;

    v8f c0 = {};
    v8f c1 = {};

    for (int kt = 0; kt < IN_DIM; kt += 32) {
        #pragma unroll
        for (int i = 0; i < 2; ++i) {
            int slot = tid + i * 256;
            int r  = slot >> 3;
            int c4 = (slot & 7) << 2;
            float4 v = *(const float4*)(xsB + (size_t)r * IN_DIM + kt + c4);
            As[r][c4 + 0] = f2bf(v.x);
            As[r][c4 + 1] = f2bf(v.y);
            As[r][c4 + 2] = f2bf(v.z);
            As[r][c4 + 3] = f2bf(v.w);
        }
        #pragma unroll
        for (int i = 0; i < 2; ++i) {
            int slot = tid + i * 256;
            int kr = slot >> 4;
            int gc = (slot & 15) << 2;
            float4 v = *(const float4*)(wB + (size_t)(kt + kr) * G_DIM + gc);
            Bs[gc + 0][kr] = f2bf(v.x);
            Bs[gc + 1][kr] = f2bf(v.y);
            Bs[gc + 2][kr] = f2bf(v.z);
            Bs[gc + 3][kr] = f2bf(v.w);
        }
        __syncthreads();

        Frag a, b0f, b1f;
        const unsigned short* ar = &As[wm * 16 + ln][0];
        a.q[0] = *(const uint4*)(ar + kh * 16);
        a.q[1] = *(const uint4*)(ar + kh * 16 + 8);

        const unsigned short* br0 = &Bs[wn * 32 + ln][kh * 16];
        const unsigned short* br1 = &Bs[wn * 32 + 16 + ln][kh * 16];
        b0f.q[0] = *(const uint4*)(br0);
        b0f.q[1] = *(const uint4*)(br0 + 8);
        b1f.q[0] = *(const uint4*)(br1);
        b1f.q[1] = *(const uint4*)(br1 + 8);

        c0 = __builtin_amdgcn_wmma_f32_16x16x32_bf16(false, a.v, false, b0f.v,
                                                     (short)0, c0, false, false);
        c1 = __builtin_amdgcn_wmma_f32_16x16x32_bf16(false, a.v, false, b1f.v,
                                                     (short)0, c1, false, false);
        __syncthreads();
    }

    const float biasA = b_ih[(size_t)b * G_DIM + g0 + wn * 32 + ln];
    const float biasB = b_ih[(size_t)b * G_DIM + g0 + wn * 32 + 16 + ln];
    float* outB = ih_all + ((size_t)b * T_DIM + t0) * G_DIM + g0;
    #pragma unroll
    for (int r = 0; r < 8; ++r) {
        int m = wm * 16 + kh * 8 + r;
        outB[(size_t)m * G_DIM + wn * 32 + ln]      = c0[r] + biasA;
        outB[(size_t)m * G_DIM + wn * 32 + 16 + ln] = c1[r] + biasB;
    }
}

// ---------------------------------------------------------------------------
// Kernel 2: pack w_hh (f32) -> bf16 pairs (one u32 holds gates g,g+1 for a k).
// Shrinks the scan's L2 working set to 100 MB (fits the 192 MB L2) and
// halves per-step L2 traffic.
// ---------------------------------------------------------------------------
__global__ __launch_bounds__(256) void gru_whh_pack(
    const float* __restrict__ w_hh, unsigned* __restrict__ whh_bf)
{
    size_t i = (size_t)blockIdx.x * blockDim.x + threadIdx.x;   // pair index
    const size_t npairs = (size_t)B_DIM * H_DIM * (G_DIM / 2);
    if (i >= npairs) return;
    float2 v = *(const float2*)(w_hh + i * 2);
    whh_bf[i] = (unsigned)f2bf(v.x) | ((unsigned)f2bf(v.y) << 16);
}

// ---------------------------------------------------------------------------
// Kernel 3: sequential GRU scan. One block per batch, 768 threads,
// 2 gate columns per thread (one bf16x2 dword per k). h lives in LDS.
// outputs: non-temporal stores; ih_all: non-temporal loads (stream-once data
// must not evict the L2-resident w_hh working set).
// ---------------------------------------------------------------------------
__global__ __launch_bounds__(768) void gru_scan(
    const float* __restrict__ state, const unsigned* __restrict__ whh_bf,
    const float* __restrict__ b_hh,  const float* __restrict__ ih_all,
    float* __restrict__ outputs, float* __restrict__ final_state)
{
    __shared__ float hs[H_DIM];
    __shared__ float hh[G_DIM];

    const int b   = blockIdx.x;
    const int tid = threadIdx.x;
    const int g0  = tid * 2;

    // pair-packed layout: [b][k][pair], pair = g0/2 = tid
    const unsigned* wb = whh_bf + (size_t)b * H_DIM * (G_DIM / 2) + tid;
    const float bias0 = b_hh[(size_t)b * G_DIM + g0];
    const float bias1 = b_hh[(size_t)b * G_DIM + g0 + 1];

    if (tid < H_DIM) hs[tid] = state[(size_t)b * H_DIM + tid];
    __syncthreads();

    for (int t = 0; t < T_DIM; ++t) {
        float acc0 = bias0, acc1 = bias1;
        #pragma unroll 8
        for (int k = 0; k < H_DIM; ++k) {
            float    hv = hs[k];                       // LDS broadcast
            unsigned w  = wb[(size_t)k * (G_DIM / 2)]; // coalesced b32
            acc0 = fmaf(hv, __uint_as_float(w << 16),        acc0);
            acc1 = fmaf(hv, __uint_as_float(w & 0xffff0000u), acc1);
        }
        hh[g0]     = acc0;
        hh[g0 + 1] = acc1;
        __syncthreads();

        if (tid < H_DIM) {
            const float* ihp = ih_all + ((size_t)b * T_DIM + t) * G_DIM;
            float i_r = __builtin_nontemporal_load(ihp + tid);
            float i_i = __builtin_nontemporal_load(ihp + H_DIM + tid);
            float i_n = __builtin_nontemporal_load(ihp + 2 * H_DIM + tid);
            float r = 1.f / (1.f + __expf(-(i_r + hh[tid])));
            float z = 1.f / (1.f + __expf(-(i_i + hh[H_DIM + tid])));
            float n = tanhf(i_n + r * hh[2 * H_DIM + tid]);
            float hprev = hs[tid];
            float hy = n + z * (hprev - n);
            __builtin_nontemporal_store(hy, outputs + ((size_t)b * T_DIM + t) * H_DIM + tid);
            hs[tid] = hy;
        }
        __syncthreads();
    }

    if (tid < H_DIM) final_state[(size_t)b * H_DIM + tid] = hs[tid];
}

extern "C" void kernel_launch(void* const* d_in, const int* in_sizes, int n_in,
                              void* d_out, int out_size, void* d_ws, size_t ws_size,
                              hipStream_t stream)
{
    (void)in_sizes; (void)n_in; (void)out_size; (void)ws_size;
    const float* xs    = (const float*)d_in[0];
    const float* state = (const float*)d_in[1];
    const float* w_ih  = (const float*)d_in[2];
    const float* w_hh  = (const float*)d_in[3];
    const float* b_ih  = (const float*)d_in[4];
    const float* b_hh  = (const float*)d_in[5];

    float* out      = (float*)d_out;
    float* final_st = out + (size_t)B_DIM * T_DIM * H_DIM;

    // workspace layout: [ih_all f32 : 402.7 MB][whh_bf16 pairs : 100.7 MB]
    float*    ih_all = (float*)d_ws;
    unsigned* whh_bf = (unsigned*)((char*)d_ws +
                        (size_t)B_DIM * T_DIM * G_DIM * sizeof(float));

    const size_t npairs = (size_t)B_DIM * H_DIM * (G_DIM / 2);
    gru_whh_pack<<<(unsigned)((npairs + 255) / 256), 256, 0, stream>>>(w_hh, whh_bf);

    dim3 grid(G_DIM / 64, T_DIM / 64, B_DIM);   // (24, 16, 64)
    gru_ih_gemm<<<grid, 256, 0, stream>>>(xs, w_ih, b_ih, ih_all);

    gru_scan<<<B_DIM, 768, 0, stream>>>(state, whh_bf, b_hh, ih_all, out, final_st);
}
